// Model_28260884808343
// MI455X (gfx1250) — matrix operand order repaired and checked by reading, not compiled, not run on hardware
//
#include <hip/hip_runtime.h>
#include <hip/hip_bf16.h>

// ---------------------------------------------------------------------------
// Problem: out[b,t,o] = sum_{j=0..10} sum_{d=0..511}
//              feat[b, clamp(t+5-j, 0, 1999), d] * W[o, j*512+d]  + bias[o]
// GEMM view: M = 32000 (b*t), N = 41 (pad to 48), K = 5632.
// Strategy: split-bf16 (hi+lo) WMMA, f32 accumulate:
//   A*B ~= Ah*Bh + Ah*Bl + Al*Bh   (3x v_wmma_f32_16x16x32_bf16)
// ---------------------------------------------------------------------------

typedef __attribute__((ext_vector_type(16))) __bf16 v16bf;
typedef __attribute__((ext_vector_type(8)))  float  v8f;

#define T_DIM      2000
#define D_DIM      512
#define K_DIM      5632          // 11 * 512
#define N_PAD      48            // 41 padded to 3 x 16
#define N_REAL     41
#define A_ROWS     26            // t0-5 .. t0+20
#define A_STRIDE   528           // 512 + 16 bf16 pad (bank spread)
#define WROW       K_DIM         // bf16 W row length

__device__ __forceinline__ unsigned short f32_to_bf16_rne(float x) {
    unsigned int u = __float_as_uint(x);
    u += 0x7FFFu + ((u >> 16) & 1u);          // round-to-nearest-even
    return (unsigned short)(u >> 16);
}
__device__ __forceinline__ float bf16_to_f32(unsigned short h) {
    return __uint_as_float(((unsigned int)h) << 16);
}

// ---- Pre-kernel: convert W (41x5632 f32) -> hi/lo bf16, padded to 48 rows --
__global__ void wconv_kernel(const float* __restrict__ W,
                             unsigned short* __restrict__ whi,
                             unsigned short* __restrict__ wlo) {
    int i = blockIdx.x * 256 + threadIdx.x;
    if (i >= N_PAD * K_DIM) return;
    int n = i / K_DIM;
    int k = i - n * K_DIM;
    float x = (n < N_REAL) ? W[n * K_DIM + k] : 0.0f;
    unsigned short h = f32_to_bf16_rne(x);
    whi[i] = h;
    wlo[i] = f32_to_bf16_rne(x - bf16_to_f32(h));
}

// A fragment (16x32 bf16, ISA layout): lane = M-row (mod 16),
// two b128 at byte +0 and +32 (K 0..7/8..15 lo tile, 16..23/24..31 hi tile).
__device__ __forceinline__ v16bf load_frag_a(const unsigned short* base) {
    union { uint4 u[2]; v16bf v; } t;
    const uint4* p = (const uint4*)base;
    t.u[0] = p[0];
    t.u[1] = p[2];
    return t.v;
}
// B fragment (32x16 bf16): lane = N-col (mod 16), two b128 at byte +0 and +32 (same K order as A).
__device__ __forceinline__ v16bf load_frag_b(const unsigned short* base) {
    union { uint4 u[2]; v16bf v; } t;
    const uint4* p = (const uint4*)base;
    t.u[0] = p[0];
    t.u[1] = p[2];
    return t.v;
}

__global__ __launch_bounds__(128)
void Model_28260884808343_kernel(const float* __restrict__ feat,
                                 const unsigned short* __restrict__ whi,
                                 const unsigned short* __restrict__ wlo,
                                 const float* __restrict__ bias,
                                 float* __restrict__ out) {
    // LDS: A tile hi/lo (2 * 26 * 528 bf16 = 54912 B); reduction reuses it.
    __shared__ alignas(16) unsigned char smem[2 * A_ROWS * A_STRIDE * 2];
    unsigned short* aHi = (unsigned short*)smem;
    unsigned short* aLo = aHi + A_ROWS * A_STRIDE;

    const int tid = threadIdx.x;
    const int b   = blockIdx.x / 125;
    const int t0  = (blockIdx.x - b * 125) * 16;

    // ---- Stage 26 clamped feature rows into LDS as hi/lo bf16 -------------
    {
        const int col = tid * 4;                      // 128 threads x 4 f32 = 512
        for (int lr = 0; lr < A_ROWS; ++lr) {
            int ts = t0 - 5 + lr;
            ts = ts < 0 ? 0 : (ts > T_DIM - 1 ? T_DIM - 1 : ts);
            const float4 f = *(const float4*)(feat +
                ((size_t)b * T_DIM + ts) * D_DIM + col);
            unsigned short h0 = f32_to_bf16_rne(f.x);
            unsigned short h1 = f32_to_bf16_rne(f.y);
            unsigned short h2 = f32_to_bf16_rne(f.z);
            unsigned short h3 = f32_to_bf16_rne(f.w);
            unsigned short l0 = f32_to_bf16_rne(f.x - bf16_to_f32(h0));
            unsigned short l1 = f32_to_bf16_rne(f.y - bf16_to_f32(h1));
            unsigned short l2 = f32_to_bf16_rne(f.z - bf16_to_f32(h2));
            unsigned short l3 = f32_to_bf16_rne(f.w - bf16_to_f32(h3));
            uint2 hv, lv;
            hv.x = (unsigned)h0 | ((unsigned)h1 << 16);
            hv.y = (unsigned)h2 | ((unsigned)h3 << 16);
            lv.x = (unsigned)l0 | ((unsigned)l1 << 16);
            lv.y = (unsigned)l2 | ((unsigned)l3 << 16);
            *(uint2*)(aHi + lr * A_STRIDE + col) = hv;
            *(uint2*)(aLo + lr * A_STRIDE + col) = lv;
        }
    }
    __syncthreads();

    // ---- Per-wave WMMA compute --------------------------------------------
    const int w     = tid >> 5;          // wave 0..3
    const int L     = tid & 31;          // lane
    const int lmod  = L & 15;
    const int lhalf = L >> 4;

    v8f acc[3] = {};                     // 3 N-tiles x 16x16 f32

    // Wave w handles offsets j = w, w+4, w+8  (11 offsets over 4 waves)
    for (int j = w; j < 11; j += 4) {
        // output row r uses feat time t0+r+5-j -> LDS row r + (10-j)
        const int aRowOff = ((10 - j) + lmod) * A_STRIDE + lhalf * 8;
        const int kColW   = j * D_DIM;                   // K base in W
        const unsigned short* aHiBase = aHi + aRowOff;
        const unsigned short* aLoBase = aLo + aRowOff;

        for (int kk = 0; kk < 16; ++kk) {                // 16 * 32 = 512 d
            const v16bf ah = load_frag_a(aHiBase + kk * 32);
            const v16bf al = load_frag_a(aLoBase + kk * 32);
            const int kb = kColW + kk * 32 + lhalf * 8;
#pragma unroll
            for (int nt = 0; nt < 3; ++nt) {
                const size_t boff = (size_t)(nt * 16 + lmod) * WROW + kb;
                const v16bf bh = load_frag_b(whi + boff);
                const v16bf bl = load_frag_b(wlo + boff);
                acc[nt] = __builtin_amdgcn_wmma_f32_16x16x32_bf16(
                    false, ah, false, bh, (short)0, acc[nt], false, false);
                acc[nt] = __builtin_amdgcn_wmma_f32_16x16x32_bf16(
                    false, ah, false, bl, (short)0, acc[nt], false, false);
                acc[nt] = __builtin_amdgcn_wmma_f32_16x16x32_bf16(
                    false, al, false, bh, (short)0, acc[nt], false, false);
            }
        }
    }

    // ---- Deterministic cross-wave reduction (fixed order) -----------------
    __syncthreads();                     // waves done reading A tile
    float* red = (float*)smem;           // 4 waves * 3 tiles * 8 vgprs * 32 lanes
#pragma unroll
    for (int nt = 0; nt < 3; ++nt)
#pragma unroll
        for (int v = 0; v < 8; ++v)
            red[(((w * 3 + nt) << 3) + v) * 32 + L] = acc[nt][v];
    __syncthreads();

    // C layout: lane = (n&15) + 16*(r>>3), vgpr = r&7  (ISA 16x16 f32 C/D)
    for (int idx = tid; idx < 16 * N_PAD; idx += 128) {
        const int r = idx / N_PAD;
        const int n = idx - r * N_PAD;
        if (n < N_REAL) {
            const int nt   = n >> 4;
            const int v    = r & 7;
            const int lane = (n & 15) + ((r >> 3) << 4);
            float s = 0.0f;
#pragma unroll
            for (int ww = 0; ww < 4; ++ww)
                s += red[(((ww * 3 + nt) << 3) + v) * 32 + lane];
            out[((size_t)b * T_DIM + t0 + r) * N_REAL + n] = s + bias[n];
        }
    }
}

extern "C" void kernel_launch(void* const* d_in, const int* in_sizes, int n_in,
                              void* d_out, int out_size, void* d_ws, size_t ws_size,
                              hipStream_t stream) {
    const float* feat = (const float*)d_in[0];   // 16*2000*512 f32
    const float* W    = (const float*)d_in[1];   // 41*5632 f32
    const float* bias = (const float*)d_in[2];   // 41 f32
    float* out        = (float*)d_out;           // 16*2000*41 f32

    // Workspace: Whi + Wlo as [48][5632] bf16 each (~1.06 MB total)
    unsigned short* whi = (unsigned short*)d_ws;
    unsigned short* wlo = whi + (size_t)N_PAD * K_DIM;

    const int welems = N_PAD * K_DIM;
    wconv_kernel<<<(welems + 255) / 256, 256, 0, stream>>>(W, whi, wlo);

    // 2000 M-tiles of 16 rows (2000 = 16 batches * 125 tiles), 4 waves each.
    Model_28260884808343_kernel<<<2000, 128, 0, stream>>>(feat, whi, wlo, bias, out);
}
